// MultiModalAttentionV2_85641647882456
// MI455X (gfx1250) — matrix operand order repaired and checked by reading, not compiled, not run on hardware
//
#include <hip/hip_runtime.h>

typedef unsigned short u16;
typedef __attribute__((ext_vector_type(16))) __bf16 bf16x16;
typedef __attribute__((ext_vector_type(8)))  float  f32x8;

#define DEV __device__ __forceinline__

// ---------- bf16 helpers (raw u16 storage) ----------
DEV u16 f2bf(float f) {
  unsigned u = __float_as_uint(f);
  unsigned r = u + 0x7FFFu + ((u >> 16) & 1u);
  return (u16)(r >> 16);
}
DEV float bf2f(u16 h) { return __uint_as_float(((unsigned)h) << 16); }

// ---------- WMMA fragment loaders ----------
// A 16x32 bf16 (row-major source): lane l -> row (l&15); lanes<16 hold K {0..7,16..23},
// lanes>=16 hold K {8..15,24..31}  (per CDNA5 ISA 7.12.2 table)
DEV bf16x16 load_a(const u16* base, int ld, int r0, int k0) {
  int lane = threadIdx.x & 31;
  int r = lane & 15;
  int kc = (lane >> 4) * 8;
  const u16* p = base + (size_t)(r0 + r) * ld + k0 + kc;
  union { bf16x16 v; uint4 q[2]; } u;
  u.q[0] = *(const uint4*)p;
  u.q[1] = *(const uint4*)(p + 16);
  return u.v;
}
// B 32x16 bf16 fed as N-major rows (B^T row-major, ld = K stride):
// lane l -> column (l&15); lanes<16 hold K {0..7,16..23}, lanes>=16 hold K {8..15,24..31}
DEV bf16x16 load_b(const u16* base, int ld, int n0, int k0) {
  int lane = threadIdx.x & 31;
  int n = lane & 15;
  int kc = (lane >> 4) * 8;
  const u16* p = base + (size_t)(n0 + n) * ld + k0 + kc;
  union { bf16x16 v; uint4 q[2]; } u;
  u.q[0] = *(const uint4*)p;
  u.q[1] = *(const uint4*)(p + 16);
  return u.v;
}
DEV f32x8 wmma_bf(bf16x16 a, bf16x16 b, f32x8 c) {
  return __builtin_amdgcn_wmma_f32_16x16x32_bf16(false, a, false, b, (short)0, c, false, false);
}

// ---------- elementwise / pack kernels ----------
__global__ void f32_to_bf16_k(const float* __restrict__ s, u16* __restrict__ d, long n) {
  long i = (long)blockIdx.x * blockDim.x + threadIdx.x;
  if (i < n) d[i] = f2bf(s[i]);
}
// W (Kd x Nd) f32 -> Wt (Nd x Kd) bf16
__global__ void transpose_bf16_k(const float* __restrict__ W, u16* __restrict__ Wt, int Kd, int Nd) {
  long i = (long)blockIdx.x * blockDim.x + threadIdx.x;
  if (i >= (long)Kd * Nd) return;
  int k = (int)(i % Kd), n = (int)(i / Kd);
  Wt[i] = f2bf(W[(size_t)k * Nd + n]);
}
// qxw[z,t,d] = q_x[b,t,h*64+d] * w4xy[h,d] * (1/8)     (z = b*8+h)
__global__ void pack_qxw_k(const u16* __restrict__ qkvx, const float* __restrict__ w4xy,
                           u16* __restrict__ qxw) {
  long i = (long)blockIdx.x * blockDim.x + threadIdx.x;
  int d = (int)(i & 63), t = (int)((i >> 6) & 1023), z = (int)(i >> 16);
  int b = z >> 3, h = z & 7;
  float v = bf2f(qkvx[((size_t)(b * 1024 + t)) * 1536 + h * 64 + d]) * w4xy[h * 64 + d] * 0.125f;
  qxw[i] = f2bf(v);
}
// vT[z,d,t] = qkv[b, t, colOff + h*64 + d]   (L = tokens per batch)
__global__ void pack_vT_k(const u16* __restrict__ qkv, u16* __restrict__ vT, int L, int colOff) {
  long i = (long)blockIdx.x * blockDim.x + threadIdx.x;
  int t = (int)(i % L); long rest = i / L;
  int d = (int)(rest & 63), z = (int)(rest >> 6);
  int b = z >> 3, h = z & 7;
  vT[i] = qkv[((size_t)(b * L + t)) * 1536 + colOff + h * 64 + d];
}
// out[z*L + t] = sum_d q[b,t,h*64+d] * w4[h,d]
__global__ void attn_bias_k(const u16* __restrict__ qkv, const float* __restrict__ w4,
                            float* __restrict__ out, int L) {
  long i = (long)blockIdx.x * blockDim.x + threadIdx.x;
  int t = (int)(i % L); int z = (int)(i / L);
  int b = z >> 3, h = z & 7;
  const u16* q = qkv + ((size_t)(b * L + t)) * 1536 + h * 64;
  const float* wv = w4 + h * 64;
  float s = 0.f;
#pragma unroll 8
  for (int d = 0; d < 64; ++d) s += bf2f(q[d]) * wv[d];
  out[i] = s;
}
__global__ void combine1_k(const float* __restrict__ cx2y, const float* __restrict__ cy2x,
                           const float* __restrict__ x, const float* __restrict__ sval,
                           float* __restrict__ cval, u16* __restrict__ cval_bf,
                           u16* __restrict__ sval_bf, long n) {
  long i = (long)blockIdx.x * blockDim.x + threadIdx.x;
  if (i >= n) return;
  float cv = cx2y[i] + cy2x[i] + x[i];
  cval[i] = cv; cval_bf[i] = f2bf(cv); sval_bf[i] = f2bf(sval[i]);
}
__global__ void combine2_k(const float* __restrict__ sg, const float* __restrict__ cg,
                           const float* __restrict__ cval, const float* __restrict__ sval,
                           u16* __restrict__ zbf, long n) {
  long i = (long)blockIdx.x * blockDim.x + threadIdx.x;
  if (i >= n) return;
  zbf[i] = f2bf(sg[i] * cval[i] + cg[i] * sval[i]);
}

// ---------- softmax kernels ----------
__global__ __launch_bounds__(128) void softmax_row_k(const float* __restrict__ S, u16* __restrict__ O) {
  int row = blockIdx.x * 4 + (threadIdx.x >> 5);
  int lane = threadIdx.x & 31;
  const float* pr = S + (size_t)row * 256;
  float v[8]; float mx = -3e38f;
#pragma unroll
  for (int i = 0; i < 8; ++i) { v[i] = pr[i * 32 + lane]; mx = fmaxf(mx, v[i]); }
  for (int off = 16; off; off >>= 1) mx = fmaxf(mx, __shfl_xor(mx, off, 32));
  float s = 0.f;
#pragma unroll
  for (int i = 0; i < 8; ++i) { v[i] = __expf(v[i] - mx); s += v[i]; }
  for (int off = 16; off; off >>= 1) s += __shfl_xor(s, off, 32);
  float inv = 1.f / s;
  u16* po = O + (size_t)row * 256;
#pragma unroll
  for (int i = 0; i < 8; ++i) po[i * 32 + lane] = f2bf(v[i] * inv);
}
// softmax over T (1024) down each column m of catt[z] (ld 256)
__global__ __launch_bounds__(128) void softmax_col_k(const float* __restrict__ S, u16* __restrict__ O) {
  int col = blockIdx.x * 4 + (threadIdx.x >> 5);
  int z = col >> 8, m = col & 255;
  int lane = threadIdx.x & 31;
  const float* p = S + (size_t)z * 1024 * 256 + m;
  float v[32]; float mx = -3e38f;
#pragma unroll
  for (int i = 0; i < 32; ++i) { v[i] = p[(size_t)(i * 32 + lane) * 256]; mx = fmaxf(mx, v[i]); }
  for (int off = 16; off; off >>= 1) mx = fmaxf(mx, __shfl_xor(mx, off, 32));
  float s = 0.f;
#pragma unroll
  for (int i = 0; i < 32; ++i) { v[i] = __expf(v[i] - mx); s += v[i]; }
  for (int off = 16; off; off >>= 1) s += __shfl_xor(s, off, 32);
  float inv = 1.f / s;
  u16* o = O + (size_t)z * 1024 * 256 + m;
#pragma unroll
  for (int i = 0; i < 32; ++i) o[(size_t)(i * 32 + lane) * 256] = f2bf(v[i] * inv);
}

// ---------- generic batched WMMA GEMM  D = A(MxK) @ Bt(NxK)^T  ----------
// Wave tile 64x32 (4 A-frags x 2 B-frags -> 8 WMMA per 6 loads);
// block = 4 waves (2x2) -> 128x64 block tile. M % 128 == 0, N % 64 == 0, K % 32 == 0.
// EPI: 0 = +bias -> bf16 | 1 = *alpha + rowAdd[z,m] + colAdd[z,n] -> f32
//      2 = f32 | 3 = +bias, sigmoid -> f32 | 4 = +bias -> f32
template <int EPI>
__global__ __launch_bounds__(128)
void gemm_k(const u16* __restrict__ A, int lda, long sAb, long sAh,
            const u16* __restrict__ Bt, int ldb, long sBb, long sBh,
            float* __restrict__ Df, u16* __restrict__ Dbf, int ldd, long sDb, long sDh,
            int M, int N, int K, int Hn,
            const float* __restrict__ bias, const float* __restrict__ rowAdd,
            const float* __restrict__ colAdd, float alpha) {
  int z = blockIdx.z, zb = z / Hn, zh = z % Hn;
  const u16* Ab = A + (size_t)zb * sAb + (size_t)zh * sAh;
  const u16* Bb = Bt + (size_t)zb * sBb + (size_t)zh * sBh;
  int w = threadIdx.x >> 5;
  int m0 = blockIdx.y * 128 + (w >> 1) * 64;
  int n0 = blockIdx.x * 64 + (w & 1) * 32;
  f32x8 c00 = {}, c01 = {}, c10 = {}, c11 = {};
  f32x8 c20 = {}, c21 = {}, c30 = {}, c31 = {};
  for (int k = 0; k < K; k += 32) {
    bf16x16 a0 = load_a(Ab, lda, m0,      k);
    bf16x16 a1 = load_a(Ab, lda, m0 + 16, k);
    bf16x16 a2 = load_a(Ab, lda, m0 + 32, k);
    bf16x16 a3 = load_a(Ab, lda, m0 + 48, k);
    bf16x16 b0 = load_b(Bb, ldb, n0,      k);
    bf16x16 b1 = load_b(Bb, ldb, n0 + 16, k);
    c00 = wmma_bf(a0, b0, c00); c01 = wmma_bf(a0, b1, c01);
    c10 = wmma_bf(a1, b0, c10); c11 = wmma_bf(a1, b1, c11);
    c20 = wmma_bf(a2, b0, c20); c21 = wmma_bf(a2, b1, c21);
    c30 = wmma_bf(a3, b0, c30); c31 = wmma_bf(a3, b1, c31);
  }
  int lane = threadIdx.x & 31, half = lane >> 4, nI = lane & 15;
  size_t dbase = (size_t)zb * sDb + (size_t)zh * sDh;
  auto epi = [&](f32x8& cc, int mt, int nt) {
#pragma unroll
    for (int j = 0; j < 8; ++j) {
      int m = m0 + mt * 16 + j + half * 8;
      int n = n0 + nt * 16 + nI;
      float v = cc[j];
      size_t off = dbase + (size_t)m * ldd + n;
      if (EPI == 0)      { v += bias[n]; Dbf[off] = f2bf(v); }
      else if (EPI == 1) { Df[off] = v * alpha + rowAdd[(size_t)z * M + m] + colAdd[(size_t)z * N + n]; }
      else if (EPI == 2) { Df[off] = v; }
      else if (EPI == 3) { v += bias[n]; Df[off] = 1.f / (1.f + __expf(-v)); }
      else               { v += bias[n]; Df[off] = v; }
    }
  };
  epi(c00, 0, 0); epi(c01, 0, 1);
  epi(c10, 1, 0); epi(c11, 1, 1);
  epi(c20, 2, 0); epi(c21, 2, 1);
  epi(c30, 3, 0); epi(c31, 3, 1);
}

// ---------- causal flash attention: one wave = 16 query rows ----------
// S = scale * Q @ K^T (causal), O = softmax(S) @ V ;  Vt is (z, 64, 1024)
template <int DK>
__global__ __launch_bounds__(128)
void flash_k(const u16* __restrict__ Q, int ldq, long sQb, long sQh,
             const u16* __restrict__ Km, int ldk, long sKb, long sKh,
             const u16* __restrict__ Vt, long sVz,
             float* __restrict__ O, int ldo, long sOb, long sOh, float scale) {
  __shared__ u16 psh[4][16][32];
  int z = blockIdx.z, zb = z >> 3, zh = z & 7;
  int w = threadIdx.x >> 5;
  int lane = threadIdx.x & 31, half = lane >> 4, nI = lane & 15;
  int t0 = (blockIdx.y * 4 + w) * 16;
  const u16* Qp = Q + (size_t)zb * sQb + (size_t)zh * sQh;
  const u16* Kp = Km + (size_t)zb * sKb + (size_t)zh * sKh;
  const u16* Vp = Vt + (size_t)z * sVz;
  float mrow[8], lrow[8];
  f32x8 o0 = {}, o1 = {}, o2 = {}, o3 = {};
#pragma unroll
  for (int j = 0; j < 8; ++j) { mrow[j] = -3e38f; lrow[j] = 0.f; }
  int nblk = (t0 + 15) / 32 + 1;
  for (int bk = 0; bk < nblk; ++bk) {
    int s0 = bk * 32;
    f32x8 sc0 = {}, sc1 = {};
#pragma unroll
    for (int kk = 0; kk < DK; kk += 32) {
      bf16x16 aq = load_a(Qp, ldq, t0, kk);
      bf16x16 b0 = load_b(Kp, ldk, s0, kk);
      bf16x16 b1 = load_b(Kp, ldk, s0 + 16, kk);
      sc0 = wmma_bf(aq, b0, sc0);
      sc1 = wmma_bf(aq, b1, sc1);
    }
#pragma unroll
    for (int j = 0; j < 8; ++j) {
      int trow = t0 + j + half * 8;
      float v0 = sc0[j] * scale, v1 = sc1[j] * scale;
      if (s0 + nI > trow)      v0 = -3e38f;
      if (s0 + 16 + nI > trow) v1 = -3e38f;
      float mx = fmaxf(v0, v1);
      for (int off = 8; off; off >>= 1) mx = fmaxf(mx, __shfl_xor(mx, off, 32));
      float mnew = fmaxf(mrow[j], mx);
      float corr = __expf(mrow[j] - mnew);
      float p0 = __expf(v0 - mnew), p1 = __expf(v1 - mnew);
      float rs = p0 + p1;
      for (int off = 8; off; off >>= 1) rs += __shfl_xor(rs, off, 32);
      lrow[j] = lrow[j] * corr + rs;
      mrow[j] = mnew;
      o0[j] *= corr; o1[j] *= corr; o2[j] *= corr; o3[j] *= corr;
      int r = j + half * 8;
      psh[w][r][nI]      = f2bf(p0);
      psh[w][r][nI + 16] = f2bf(p1);
    }
    bf16x16 pa = load_a(&psh[w][0][0], 32, 0, 0);
    o0 = wmma_bf(pa, load_b(Vp, 1024, 0,  s0), o0);
    o1 = wmma_bf(pa, load_b(Vp, 1024, 16, s0), o1);
    o2 = wmma_bf(pa, load_b(Vp, 1024, 32, s0), o2);
    o3 = wmma_bf(pa, load_b(Vp, 1024, 48, s0), o3);
  }
#pragma unroll
  for (int j = 0; j < 8; ++j) {
    float inv = 1.f / lrow[j];
    int trow = t0 + j + half * 8;
    float* op = O + (size_t)zb * sOb + (size_t)zh * sOh + (size_t)trow * ldo;
    op[nI]      = o0[j] * inv;
    op[16 + nI] = o1[j] * inv;
    op[32 + nI] = o2[j] * inv;
    op[48 + nI] = o3[j] * inv;
  }
}

// ---------- host launch ----------
extern "C" void kernel_launch(void* const* d_in, const int* in_sizes, int n_in,
                              void* d_out, int out_size, void* d_ws, size_t ws_size,
                              hipStream_t stream) {
  const float* x      = (const float*)d_in[0];
  const float* y      = (const float*)d_in[1];
  const float* Wqkv_x = (const float*)d_in[3];
  const float* bqkv_x = (const float*)d_in[4];
  const float* Wqkv_y = (const float*)d_in[5];
  const float* bqkv_y = (const float*)d_in[6];
  const float* w4x    = (const float*)d_in[7];
  const float* w4y    = (const float*)d_in[8];
  const float* w4xy   = (const float*)d_in[9];
  const float* Wgs    = (const float*)d_in[10];
  const float* bgs    = (const float*)d_in[11];
  const float* Wgc    = (const float*)d_in[12];
  const float* bgc    = (const float*)d_in[13];
  const float* Wp     = (const float*)d_in[14];
  const float* bp     = (const float*)d_in[15];

  char* p = (char*)d_ws;
  auto alloc = [&](size_t bytes) -> void* {
    void* r = (void*)p; p += (bytes + 255) & ~(size_t)255; return r;
  };
  const long BT = 4096, BM = 1024, C = 512, C3 = 1536, Z = 32, T = 1024, M = 256;

  u16* Xbf    = (u16*)alloc(BT * C * 2);
  u16* Ybf    = (u16*)alloc(BM * C * 2);
  u16* WqkvxT = (u16*)alloc(C3 * C * 2);
  u16* WqkvyT = (u16*)alloc(C3 * C * 2);
  u16* WgsT   = (u16*)alloc(C * C * 2);
  u16* WgcT   = (u16*)alloc(C * C * 2);
  u16* WpT    = (u16*)alloc(C * C * 2);
  u16* qkvx   = (u16*)alloc(BT * C3 * 2);
  u16* qkvy   = (u16*)alloc(BM * C3 * 2);
  u16* qxw    = (u16*)alloc(Z * T * 64 * 2);
  u16* vxT    = (u16*)alloc(Z * 64 * T * 2);
  u16* vyT    = (u16*)alloc(Z * 64 * M * 2);
  float* catt  = (float*)alloc(Z * T * M * 4);
  float* catt1 = (float*)alloc(Z * T * 4);
  float* catt2 = (float*)alloc(Z * M * 4);
  u16* cx2y   = (u16*)alloc(Z * T * M * 2);
  u16* cy2x   = (u16*)alloc(Z * T * M * 2);
  float* cvx2y = (float*)alloc(BT * C * 4);
  float* cvy2x = (float*)alloc(BT * C * 4);
  float* sval  = (float*)alloc(BT * C * 4);
  float* cval  = (float*)alloc(BT * C * 4);
  u16* cval_bf = (u16*)alloc(BT * C * 2);
  u16* sval_bf = (u16*)alloc(BT * C * 2);
  float* sgate = (float*)alloc(BT * C * 4);
  float* cgate = (float*)alloc(BT * C * 4);
  u16* zbf     = (u16*)alloc(BT * C * 2);

  // 1. packs
  f32_to_bf16_k<<<(BT * C) / 256, 256, 0, stream>>>(x, Xbf, BT * C);
  f32_to_bf16_k<<<(BM * C) / 256, 256, 0, stream>>>(y, Ybf, BM * C);
  transpose_bf16_k<<<(C * C3) / 256, 256, 0, stream>>>(Wqkv_x, WqkvxT, C, C3);
  transpose_bf16_k<<<(C * C3) / 256, 256, 0, stream>>>(Wqkv_y, WqkvyT, C, C3);
  transpose_bf16_k<<<(C * C) / 256, 256, 0, stream>>>(Wgs, WgsT, C, C);
  transpose_bf16_k<<<(C * C) / 256, 256, 0, stream>>>(Wgc, WgcT, C, C);
  transpose_bf16_k<<<(C * C) / 256, 256, 0, stream>>>(Wp, WpT, C, C);

  // 2. QKV projections (bias -> bf16)
  gemm_k<0><<<dim3(24, 32, 1), 128, 0, stream>>>(Xbf, 512, 0, 0, WqkvxT, 512, 0, 0,
      nullptr, qkvx, 1536, 0, 0, 4096, 1536, 512, 1, bqkv_x, nullptr, nullptr, 1.f);
  gemm_k<0><<<dim3(24, 8, 1), 128, 0, stream>>>(Ybf, 512, 0, 0, WqkvyT, 512, 0, 0,
      nullptr, qkvy, 1536, 0, 0, 1024, 1536, 512, 1, bqkv_y, nullptr, nullptr, 1.f);

  // 3. repacks + additive biases
  pack_qxw_k<<<(Z * T * 64) / 256, 256, 0, stream>>>(qkvx, w4xy, qxw);
  pack_vT_k<<<(Z * 64 * T) / 256, 256, 0, stream>>>(qkvx, vxT, 1024, 1024);
  pack_vT_k<<<(Z * 64 * M) / 256, 256, 0, stream>>>(qkvy, vyT, 256, 1024);
  attn_bias_k<<<(Z * T) / 256, 256, 0, stream>>>(qkvx, w4x, catt1, 1024);
  attn_bias_k<<<(Z * M) / 256, 256, 0, stream>>>(qkvy, w4y, catt2, 256);

  // 4. catt = qxw @ k_y^T + catt1 + catt2   (scale folded into qxw)
  gemm_k<1><<<dim3(4, 8, 32), 128, 0, stream>>>(
      qxw, 64, (long)8 * T * 64, (long)T * 64,
      qkvy + 512, 1536, (long)256 * 1536, 64,
      catt, nullptr, 256, (long)8 * T * 256, (long)T * 256,
      1024, 256, 64, 8, nullptr, catt1, catt2, 1.f);

  // 5. softmaxes
  softmax_row_k<<<(Z * T) / 4, 128, 0, stream>>>(catt, cx2y);
  softmax_col_k<<<(Z * M) / 4, 128, 0, stream>>>(catt, cy2x);

  // 6a. cval_x2y = cx2y @ v_y
  gemm_k<2><<<dim3(1, 8, 32), 128, 0, stream>>>(
      cx2y, 256, (long)8 * T * 256, (long)T * 256,
      vyT, 256, (long)8 * 64 * 256, (long)64 * 256,
      cvx2y, nullptr, 512, (long)T * 512, 64,
      1024, 64, 256, 8, nullptr, nullptr, nullptr, 1.f);
  // 6b. chain attention: Q=cx2y, K=cy2x (d=256), V=v_x, causal, scale 1/16
  flash_k<256><<<dim3(1, 16, 32), 128, 0, stream>>>(
      cx2y, 256, (long)8 * T * 256, (long)T * 256,
      cy2x, 256, (long)8 * T * 256, (long)T * 256,
      vxT, (long)64 * 1024,
      cvy2x, 512, (long)T * 512, 64, 0.0625f);
  // 6c. self attention: q_x, k_x, v_x, causal, scale 1/8
  flash_k<64><<<dim3(1, 16, 32), 128, 0, stream>>>(
      qkvx, 1536, (long)T * 1536, 64,
      qkvx + 512, 1536, (long)T * 1536, 64,
      vxT, (long)64 * 1024,
      sval, 512, (long)T * 512, 64, 0.125f);

  // 7. cval = cvx2y + cvy2x + x ; bf16 packs
  combine1_k<<<(BT * C) / 256, 256, 0, stream>>>(cvx2y, cvy2x, x, sval,
                                                 cval, cval_bf, sval_bf, BT * C);

  // 8. gates (sigmoid epilogue)
  gemm_k<3><<<dim3(8, 32, 1), 128, 0, stream>>>(sval_bf, 512, 0, 0, WgsT, 512, 0, 0,
      sgate, nullptr, 512, 0, 0, 4096, 512, 512, 1, bgs, nullptr, nullptr, 1.f);
  gemm_k<3><<<dim3(8, 32, 1), 128, 0, stream>>>(cval_bf, 512, 0, 0, WgcT, 512, 0, 0,
      cgate, nullptr, 512, 0, 0, 4096, 512, 512, 1, bgc, nullptr, nullptr, 1.f);

  // 9. z = sgate*cval + cgate*sval
  combine2_k<<<(BT * C) / 256, 256, 0, stream>>>(sgate, cgate, cval, sval, zbf, BT * C);

  // 10. out = z @ Wp + bp
  gemm_k<4><<<dim3(8, 32, 1), 128, 0, stream>>>(zbf, 512, 0, 0, WpT, 512, 0, 0,
      (float*)d_out, nullptr, 512, 0, 0, 4096, 512, 512, 1, bp, nullptr, nullptr, 1.f);
}
